// HLSTM_86337432584254
// MI455X (gfx1250) — matrix operand order repaired and checked by reading, not compiled, not run on hardware
//
#include <hip/hip_runtime.h>
#include <hip/hip_bf16.h>

// ---------------------------------------------------------------------------
// Problem constants: B=64, T=512, F=6, H=512, O=2
//   BT  = B*T = 32768 rows, G4 = 4*H = 2048 gate columns per direction
// ---------------------------------------------------------------------------

typedef __attribute__((ext_vector_type(16))) __bf16 v16bf;
typedef __attribute__((ext_vector_type(8)))  __bf16 v8bf;
typedef __attribute__((ext_vector_type(8)))  float  v8f;

#define WMMA_BF16(A, Bm, C) \
  __builtin_amdgcn_wmma_f32_16x16x32_bf16(false, (A), false, (Bm), (short)0, (C), false, false)

// A-matrix fragment (16x32 bf16, row-major src, leading dim lda):
// lane L<16 holds row M=L, K in {kb..kb+7, kb+16..kb+23} with kb = (L>=16)*8
static __device__ __forceinline__ v16bf load_frag_a(const __bf16* __restrict__ p,
                                                    int lda, int lane) {
  const int r  = lane & 15;
  const int kb = (lane >> 4) * 8;
  const __bf16* q = p + (size_t)r * lda + kb;
  v8bf lo = *(const v8bf*)(q);
  v8bf hi = *(const v8bf*)(q + 16);
  return __builtin_shufflevector(lo, hi, 0, 1, 2, 3, 4, 5, 6, 7,
                                 8, 9, 10, 11, 12, 13, 14, 15);
}

// B-matrix fragment (32x16 bf16). Weights stored [N, K] row-major (= B^T):
// lane L<16 takes column N=L, K in {kb..kb+7, kb+16..kb+23} with kb = (L>=16)*8.
static __device__ __forceinline__ v16bf load_frag_b(const __bf16* __restrict__ p,
                                                    int ldk, int lane) {
  const int c  = lane & 15;
  const int kb = (lane >> 4) * 8;
  const __bf16* q = p + (size_t)c * ldk + kb;
  v8bf lo = *(const v8bf*)(q);
  v8bf hi = *(const v8bf*)(q + 16);
  return __builtin_shufflevector(lo, hi, 0, 1, 2, 3, 4, 5, 6, 7,
                                 8, 9, 10, 11, 12, 13, 14, 15);
}

static __device__ __forceinline__ float sigmoidf_fast(float x) {
  return 1.0f / (1.0f + __expf(-x));
}

// --- CDNA5 async memory->LDS path (ASYNCcnt) -------------------------------
typedef __attribute__((address_space(3))) char lds_char_t;

static __device__ __forceinline__ unsigned lds_offset_of(const void* p) {
  return (unsigned)(unsigned long long)(lds_char_t*)(char*)p;
}
static __device__ __forceinline__ void async_load_b128(unsigned lds_off,
                                                       const void* gaddr) {
  asm volatile("global_load_async_to_lds_b128 %0, %1, off"
               :: "v"(lds_off), "v"(gaddr) : "memory");
}
static __device__ __forceinline__ void wait_asynccnt0() {
  asm volatile("s_wait_asynccnt 0x0" ::: "memory");
}

// ---------------------------------------------------------------------------
// fp32 -> bf16 conversion (weights, one-shot per launch)
// ---------------------------------------------------------------------------
__global__ void cvt_bf16_kernel(const float* __restrict__ src,
                                __bf16* __restrict__ dst, int n) {
  int i = blockIdx.x * blockDim.x + threadIdx.x;
  if (i < n) dst[i] = (__bf16)src[i];
}

// ---------------------------------------------------------------------------
// Encoder: act[m][j] = bf16( SELU( x[m,0:6] . enc_w[j,0:6] + enc_b[j] ) )
// ---------------------------------------------------------------------------
__global__ void encoder_kernel(const float* __restrict__ x,
                               const float* __restrict__ ew,
                               const float* __restrict__ eb,
                               __bf16* __restrict__ act) {
  int idx = blockIdx.x * blockDim.x + threadIdx.x;  // BT*512 threads
  if (idx >= 32768 * 512) return;
  int m = idx >> 9;
  int j = idx & 511;
  const float* xp = x + (size_t)m * 6;
  const float* wp = ew + (size_t)j * 6;
  float s = eb[j];
#pragma unroll
  for (int f = 0; f < 6; ++f) s += xp[f] * wp[f];
  const float scale = 1.0507009873554805f;
  const float alpha = 1.6732632423543772f;
  float r = (s > 0.0f) ? scale * s : scale * alpha * (__expf(s) - 1.0f);
  act[idx] = (__bf16)r;
}

// ---------------------------------------------------------------------------
// Input projection (time-parallel GEMM):
//   pre[d][t][b][n] = act[b*T+t][0:K] . w[d][n][0:K] + b_ih[d][n] + b_hh[d][n]
// Note pre is stored [d][t][b][n] so the recurrent kernel's per-timestep tile
// (16 batch rows x 2048 gates) is one contiguous 128 KB block.
// One 16(M)x64(N) strip per wave; grid = (2048/64, BT/128, 2), block = 256.
// ---------------------------------------------------------------------------
__global__ void proj_kernel(const __bf16* __restrict__ act, int K,
                            const __bf16* __restrict__ w,    // [2][2048][K]
                            const float* __restrict__ b_ih,  // [2][2048]
                            const float* __restrict__ b_hh,  // [2][2048]
                            float* __restrict__ pre) {       // [2][T][B][2048]
  const int lane = threadIdx.x & 31;
  const int wave = threadIdx.x >> 5;
  const int n0 = blockIdx.x * 64;
  const int m0 = blockIdx.y * 128 + wave * 16;   // BT row tile (same b)
  const int d  = blockIdx.z;
  const __bf16* wd = w + (size_t)d * 2048 * (size_t)K;
  float* pred = pre + (size_t)d * 32768 * 2048;

  const int hl = lane >> 4;
  const int cn = lane & 15;
  const int b  = m0 >> 9;         // batch index (constant over the 16-row tile)
  const int t0 = m0 & 511;        // time offset of row 0

  v8f acc[4];
#pragma unroll
  for (int nt = 0; nt < 4; ++nt) {
    const int n = n0 + nt * 16 + cn;
    float bv = b_ih[d * 2048 + n] + b_hh[d * 2048 + n];
#pragma unroll
    for (int g = 0; g < 8; ++g) acc[nt][g] = bv;
  }

  for (int k = 0; k < K; k += 32) {
    v16bf a = load_frag_a(act + (size_t)m0 * K + k, K, lane);
#pragma unroll
    for (int nt = 0; nt < 4; ++nt) {
      v16bf bf = load_frag_b(wd + (size_t)(n0 + nt * 16) * K + k, K, lane);
      acc[nt] = WMMA_BF16(a, bf, acc[nt]);
    }
  }

#pragma unroll
  for (int nt = 0; nt < 4; ++nt) {
#pragma unroll
    for (int g = 0; g < 8; ++g) {
      const int t = t0 + g + hl * 8;             // C/D layout: VGPR g, lane half
      pred[((size_t)t * 64 + b) * 2048 + n0 + nt * 16 + cn] = acc[nt][g];
    }
  }
}

// ---------------------------------------------------------------------------
// Recurrent bidirectional LSTM layer.
// One persistent workgroup per (dir, 16-row batch tile): grid = (4, 2),
// block = 512 threads (16 waves). No cross-workgroup sync.
//   - h state:  LDS bf16 [16][512] (WMMA A operand)          16 KB
//   - gates:    LDS f32  [16][2048] staging                 128 KB
//   - pre tile: LDS f32  [16][2048] async-prefetched        128 KB  (272 KB tot)
//   - c state:  registers (lane owns hidden unit j = wave*32+lane, 16 rows)
// The next timestep's pre tile streams memory->LDS with
// global_load_async_to_lds_b128 during the elementwise phase and is waited
// with s_wait_asynccnt just before the barrier — the copy hides behind the
// transcendental-heavy cell update.
// ---------------------------------------------------------------------------
__global__ void lstm_kernel(const float* __restrict__ pre,    // [2][T][B][2048]
                            const __bf16* __restrict__ whh,   // [2][2048][512]
                            float* __restrict__ out,          // [BT][1024]
                            __bf16* __restrict__ actn) {      // [BT][1024]
  extern __shared__ char smem[];
  __bf16* h_lds = (__bf16*)smem;                            // 16*512*2
  float*  g_lds = (float*)(smem + 16 * 512 * 2);            // 16*2048*4
  float*  p_lds = (float*)(smem + 16 * 512 * 2 + 16 * 2048 * 4);

  const int tid  = threadIdx.x;
  const int lane = tid & 31;
  const int wave = tid >> 5;
  const int b0 = blockIdx.x * 16;       // batch tile
  const int d  = blockIdx.y;            // direction
  const float*  pre_d = pre + (size_t)d * 32768 * 2048;
  const __bf16* whh_d = whh + (size_t)d * 2048 * 512;
  const unsigned p_off = lds_offset_of(p_lds);

  for (int i = tid; i < 16 * 512; i += blockDim.x) h_lds[i] = (__bf16)0.0f;

  float c_reg[16];
#pragma unroll
  for (int m = 0; m < 16; ++m) c_reg[m] = 0.0f;

  const int j  = wave * 32 + lane;      // hidden unit owned in elementwise phase
  const int hl = lane >> 4;
  const int cn = lane & 15;

  // Prefetch pre tile for the first timestep: 128 KB contiguous,
  // 512 threads x 16 B x 16 iterations.
  {
    const int t = d ? 511 : 0;
    const char* g = (const char*)(pre_d + ((size_t)t * 64 + b0) * 2048);
#pragma unroll
    for (int r = 0; r < 16; ++r) {
      const unsigned o = (unsigned)(tid * 16 + r * 8192);
      async_load_b128(p_off + o, g + o);
    }
  }
  wait_asynccnt0();
  __syncthreads();

  for (int s = 0; s < 512; ++s) {
    const int t = d ? (511 - s) : s;

    // ---- matmul phase: acc = h @ w_hh^T (zero-seeded) ----
    v8f acc[8];
#pragma unroll
    for (int nt = 0; nt < 8; ++nt)
#pragma unroll
      for (int g = 0; g < 8; ++g) acc[nt][g] = 0.0f;

    for (int kk = 0; kk < 512; kk += 32) {
      v16bf a = load_frag_a(h_lds + kk, 512, lane);   // h from LDS
#pragma unroll
      for (int nt = 0; nt < 8; ++nt) {
        v16bf bf = load_frag_b(whh_d + (size_t)(wave * 128 + nt * 16) * 512 + kk,
                               512, lane);
        acc[nt] = WMMA_BF16(a, bf, acc[nt]);
      }
    }
    // gates = acc + pre (same [16][2048] tile index for read and write)
#pragma unroll
    for (int nt = 0; nt < 8; ++nt) {
#pragma unroll
      for (int g = 0; g < 8; ++g) {
        const int i = (g + hl * 8) * 2048 + wave * 128 + nt * 16 + cn;
        g_lds[i] = acc[nt][g] + p_lds[i];
      }
    }
    __syncthreads();   // (1) gate-store done; p_lds reads complete

    // ---- issue async prefetch of next step's pre tile into p_lds ----
    if (s + 1 < 512) {
      const int tn = d ? (511 - (s + 1)) : (s + 1);
      const char* g = (const char*)(pre_d + ((size_t)tn * 64 + b0) * 2048);
#pragma unroll
      for (int r = 0; r < 16; ++r) {
        const unsigned o = (unsigned)(tid * 16 + r * 8192);
        async_load_b128(p_off + o, g + o);
      }
    }

    // ---- elementwise phase: LSTM cell update for hidden unit j ----
#pragma unroll
    for (int m = 0; m < 16; ++m) {
      float gi = g_lds[m * 2048 + j];
      float gf = g_lds[m * 2048 + 512 + j];
      float gg = g_lds[m * 2048 + 1024 + j];
      float go = g_lds[m * 2048 + 1536 + j];
      gi = sigmoidf_fast(gi);
      gf = sigmoidf_fast(gf);
      go = sigmoidf_fast(go);
      gg = tanhf(gg);
      float cc = gf * c_reg[m] + gi * gg;
      c_reg[m] = cc;
      float hh = go * tanhf(cc);
      h_lds[m * 512 + j] = (__bf16)hh;
      const size_t oi = ((size_t)(b0 + m) * 512 + t) * 1024 + (size_t)d * 512 + j;
      out[oi]  = hh;
      actn[oi] = (__bf16)hh;   // bf16 copy feeds next layer's projection
    }

    wait_asynccnt0();  // my slice of next pre tile is in LDS
    __syncthreads();   // (2) all waves' prefetch visible; h_lds consistent
  }
}

// ---------------------------------------------------------------------------
// Output heads: mean/disp = out @ w^T + b  (O=2; memory bound)
// ---------------------------------------------------------------------------
__global__ void head_kernel(const float* __restrict__ out2h,   // [BT][1024]
                            const float* __restrict__ mw, const float* __restrict__ mb,
                            const float* __restrict__ dw, const float* __restrict__ db,
                            float* __restrict__ dst) {         // mean | disp
  int m = blockIdx.x * blockDim.x + threadIdx.x;
  if (m >= 32768) return;
  const float* r = out2h + (size_t)m * 1024;
  float a0 = mb[0], a1 = mb[1], a2 = db[0], a3 = db[1];
  for (int k = 0; k < 1024; ++k) {
    float v = r[k];
    a0 += v * mw[k];
    a1 += v * mw[1024 + k];
    a2 += v * dw[k];
    a3 += v * dw[1024 + k];
  }
  dst[(size_t)m * 2 + 0] = a0;
  dst[(size_t)m * 2 + 1] = a1;
  dst[(size_t)32768 * 2 + (size_t)m * 2 + 0] = a2;
  dst[(size_t)32768 * 2 + (size_t)m * 2 + 1] = a3;
}

// ---------------------------------------------------------------------------
// Host-side orchestration
// ---------------------------------------------------------------------------
extern "C" void kernel_launch(void* const* d_in, const int* in_sizes, int n_in,
                              void* d_out, int out_size, void* d_ws, size_t ws_size,
                              hipStream_t stream) {
  (void)in_sizes; (void)n_in; (void)out_size; (void)ws_size;

  const float* x  = (const float*)d_in[0];
  const float* ew = (const float*)d_in[1];
  const float* eb = (const float*)d_in[2];
  const float* wih[3] = {(const float*)d_in[3], (const float*)d_in[7],  (const float*)d_in[11]};
  const float* whh[3] = {(const float*)d_in[4], (const float*)d_in[8],  (const float*)d_in[12]};
  const float* bih[3] = {(const float*)d_in[5], (const float*)d_in[9],  (const float*)d_in[13]};
  const float* bhh[3] = {(const float*)d_in[6], (const float*)d_in[10], (const float*)d_in[14]};
  const float* mw = (const float*)d_in[15];
  const float* mb = (const float*)d_in[16];
  const float* dw = (const float*)d_in[17];
  const float* db = (const float*)d_in[18];

  // Workspace layout:
  //   pre      : 2 * 32768 * 2048 f32  = 512 MiB  ([d][t][b][n], per-layer reuse)
  //   act      : 32768 * 1024 bf16     =  64 MiB
  //   outb     : 32768 * 1024 f32      = 128 MiB
  //   wih_bf16 / whh_bf16              =  32 MiB
  char* ws = (char*)d_ws;
  float*  pre    = (float*)ws;
  size_t  off    = (size_t)2 * 32768 * 2048 * 4;
  __bf16* act    = (__bf16*)(ws + off);   off += (size_t)32768 * 1024 * 2;
  float*  outb   = (float*)(ws + off);    off += (size_t)32768 * 1024 * 4;
  __bf16* wih_bf = (__bf16*)(ws + off);   off += (size_t)(2097152 + 2 * 4194304) * 2;
  __bf16* whh_bf = (__bf16*)(ws + off);

  __bf16* wih_bf_l[3] = {wih_bf, wih_bf + 2097152, wih_bf + 2097152 + 4194304};
  const int wih_n[3] = {2097152, 4194304, 4194304};

  for (int l = 0; l < 3; ++l) {
    cvt_bf16_kernel<<<(wih_n[l] + 255) / 256, 256, 0, stream>>>(wih[l], wih_bf_l[l], wih_n[l]);
    cvt_bf16_kernel<<<(2097152 + 255) / 256, 256, 0, stream>>>(
        whh[l], whh_bf + (size_t)l * 2097152, 2097152);
  }

  encoder_kernel<<<(32768 * 512) / 256, 256, 0, stream>>>(x, ew, eb, act);

  for (int l = 0; l < 3; ++l) {
    const int K = l ? 1024 : 512;
    dim3 pg(2048 / 64, 32768 / 128, 2);
    proj_kernel<<<pg, 256, 0, stream>>>(act, K, wih_bf_l[l], bih[l], bhh[l], pre);

    dim3 rg(4, 2);
    const size_t smem = (size_t)16 * 512 * 2 + 2 * (size_t)16 * 2048 * 4;  // 272 KiB
    lstm_kernel<<<rg, 512, smem, stream>>>(pre, whh_bf + (size_t)l * 2097152, outb, act);
  }

  head_kernel<<<(32768 + 255) / 256, 256, 0, stream>>>(outb, mw, mb, dw, db, (float*)d_out);
}
